// QARULayer_70231305224921
// MI455X (gfx1250) — matrix operand order repaired and checked by reading, not compiled, not run on hardware
//
#include <hip/hip_runtime.h>

// ---------------------------------------------------------------------------
// QARULayer for MI455X (gfx1250, wave32, WMMA bf16 + double-buffered TDM)
//   B=4, CIN=32, HID=32, OC=64, T=31, H=96, W=96
// ---------------------------------------------------------------------------

typedef __attribute__((ext_vector_type(16))) __bf16 v16bf;
typedef __attribute__((ext_vector_type(8)))  float  v8f;
typedef unsigned int v4u __attribute__((ext_vector_type(4)));
typedef int          v8i __attribute__((ext_vector_type(8)));
typedef int          v4i __attribute__((ext_vector_type(4)));

union BF16Frag {
    v16bf v;
    __bf16 h[16];
    uint4 q[2];
};

#define CIN 32
#define OC  64
#define HID 32
#define TT  31
#define HH  96
#define WW  96
#define NB  4

// padded bf16 input tensor: [b][33][98][98][32]  (t,y,x each padded by 1)
#define TP 33
#define HP 98
#define WP 98

#define SLAB_ELEMS (9 * 18 * CIN)     // 5184 bf16 per t-slab
#define SLAB_BYTES (SLAB_ELEMS * 2)   // 10368 B

#if __has_builtin(__builtin_amdgcn_tensor_load_to_lds)
#define HAVE_TDM 1
#endif
#if __has_builtin(__builtin_amdgcn_global_load_async_to_lds_b128)
#define HAVE_ASYNC 1
#endif
#if __has_builtin(__builtin_amdgcn_s_wait_tensorcnt)
#define WAIT_TENSOR() __builtin_amdgcn_s_wait_tensorcnt(0)
#else
#define WAIT_TENSOR() asm volatile("s_wait_tensorcnt 0x0" ::: "memory")
#endif
#define WAIT_ASYNC() asm volatile("s_wait_asynccnt 0x0" ::: "memory")

__device__ __forceinline__ unsigned lds_off_u32(const void* p) {
    return (unsigned)(unsigned long long)(__attribute__((address_space(3))) const char*)p;
}

// ---------------------------------------------------------------------------
// Kernel 0a: repack conv weights [64][32][3][3][3] f32 -> bf16 WMMA A-fragment
// order: wf[q][waveTile][lane][elem]  (q = kt*9+ky*3+kx)
// ---------------------------------------------------------------------------
__global__ __launch_bounds__(256) void pack_weights(const float* __restrict__ w,
                                                    __bf16* __restrict__ wf) {
    int idx = blockIdx.x * 256 + threadIdx.x;
    if (idx >= 27 * 4 * 32 * 16) return;
    int e    = idx & 15;
    int lane = (idx >> 4) & 31;
    int wv   = (idx >> 9) & 3;
    int q    = idx >> 11;
    int m     = lane & 15;
    int kBase = (lane >> 4) * 8;
    int p     = e >> 1;
    int k     = ((p & 4) ? 16 : 0) + kBase + ((p & 3) << 1) + (e & 1);
    int oc    = wv * 16 + m;
    int qt = q / 9, rem = q % 9, qy = rem / 3, qx = rem % 3;
    wf[idx] = (__bf16)w[(((oc * CIN + k) * 3 + qt) * 3 + qy) * 3 + qx];
}

// ---------------------------------------------------------------------------
// Kernel 0b: pad + transpose + convert input to xT[b][tp][yp][xp][cin] bf16
// ---------------------------------------------------------------------------
__global__ __launch_bounds__(256) void pad_transpose(const float* __restrict__ x,
                                                     __bf16* __restrict__ xT) {
    long idx = (long)blockIdx.x * 256 + threadIdx.x;
    const long total = (long)NB * TP * HP * WP * CIN;
    if (idx >= total) return;
    int cin = idx & 31;
    long r = idx >> 5;
    int xp = (int)(r % WP); r /= WP;
    int yp = (int)(r % HP); r /= HP;
    int tp = (int)(r % TP);
    int b  = (int)(r / TP);
    float v = 0.f;
    if (xp >= 1 && xp <= WW && yp >= 1 && yp <= HH && tp >= 1 && tp <= TT)
        v = x[(((long)(b * CIN + cin) * TT + (tp - 1)) * HH + (yp - 1)) * WW + (xp - 1)];
    xT[idx] = (__bf16)v;
}

// ---------------------------------------------------------------------------
// Kernel 1: fused conv3d (implicit GEMM bf16 WMMA) + gates + temporal scan.
// Block = (b, y, 16 x's); 4 waves = 4 oc-tiles of 16.
// Double-buffered LDS slab; TDM for slab t+1 issued before computing slab t.
// B-fragment loads software-pipelined against WMMA.
// ---------------------------------------------------------------------------
__global__ __launch_bounds__(128) void conv_gate_scan(
    const __bf16* __restrict__ xT, const __bf16* __restrict__ wf,
    const float* __restrict__ bias, float* __restrict__ hT) {

    __shared__ __align__(16) __bf16 sIn[2][SLAB_ELEMS];  // 2 x 10368 B
    __shared__ float sGates[OC][16];
    __shared__ float sH[HID][16];

    const int tid  = threadIdx.x;
    const int lane = tid & 31;
    const int wv   = tid >> 5;
    const int x0   = blockIdx.x * 16;
    const int y    = blockIdx.y;
    const int b    = blockIdx.z;

    const int n    = lane & 15;
    const int kB16 = (lane >> 4) * 8;

    for (int i = tid; i < HID * 16; i += 128) sH[i >> 4][i & 15] = 0.f;

    float zb[4], fb[4];
    #pragma unroll
    for (int k = 0; k < 4; ++k) {
        int c = (tid + k * 128) >> 4;
        zb[k] = bias[c];
        fb[k] = bias[c + HID];
    }
    const uint4* wq = reinterpret_cast<const uint4*>(wf);

#if defined(HAVE_TDM)
    // TDM descriptor groups 1..3 (t-invariant): 3D tile of 2-byte elements,
    // tile = 576 x 3(dy) x 3(dt); dim0_stride = WP*CIN, dim1_stride = HP*WP*CIN
    v8i g1;
    g1[0] = (1 << 16);                                   // data_size = 2 bytes
    g1[1] = (int)((unsigned)(WP * CIN) << 16);           // tensor_dim0 lo16
    g1[2] = (HP << 16);                                  // dim0 hi | tensor_dim1 lo16
    g1[3] = (576 << 16);                                 // dim1 hi | tile_dim0
    g1[4] = 3 | (3 << 16);                               // tile_dim1, tile_dim2
    g1[5] = WP * CIN;                                    // tensor_dim0_stride lo32
    g1[6] = (int)(((unsigned)(HP * WP * CIN) & 0xFFFFu) << 16);
    g1[7] = (int)((unsigned)(HP * WP * CIN) >> 16);
    v4i g2 = { TP, 0, 0, 0 };
    v4i g3 = { 0, 0, 0, 0 };
    const unsigned sInOff = lds_off_u32(&sIn[0][0]);

    auto tdm_issue = [&](int tt, int buf) {
        const __bf16* src0 = xT + ((((long)b * TP + tt) * HP + y) * WP + x0) * CIN;
        unsigned long ga = (unsigned long)(const void*)src0;
        v4u g0;
        g0[0] = 1u;                                            // count=1
        g0[1] = sInOff + (unsigned)(buf * SLAB_BYTES);         // lds_addr
        g0[2] = (unsigned)ga;                                  // global lo32
        g0[3] = ((unsigned)((ga >> 32) & 0x1FFFFFFu)) | (2u << 30);
#if __clang_major__ >= 23
        v8i g4 = { 0, 0, 0, 0, 0, 0, 0, 0 };
        __builtin_amdgcn_tensor_load_to_lds(g0, g1, g2, g3, g4, 0);
#else
        __builtin_amdgcn_tensor_load_to_lds(g0, g1, g2, g3, 0);
#endif
    };
    // prologue: slab for t=0
    if (wv == 0) {
        tdm_issue(0, 0);
        WAIT_TENSOR();
    }
#elif defined(HAVE_ASYNC)
    auto async_issue = [&](int tt, int buf) {
        for (int i16 = tid; i16 < 648; i16 += 128) {
            int row = i16 / 72, c16 = i16 % 72;
            int dt = row / 3, dy = row % 3;
            const __bf16* src =
                xT + ((((long)b * TP + (tt + dt)) * HP + (y + dy)) * WP + x0) * CIN + c16 * 8;
            __bf16* dst = &sIn[buf][row * 576 + c16 * 8];
            __builtin_amdgcn_global_load_async_to_lds_b128(
                (const __attribute__((address_space(1))) void*)src,
                (__attribute__((address_space(3))) void*)dst, 0, 0);
        }
    };
    async_issue(0, 0);
    WAIT_ASYNC();
#endif
    __syncthreads();

    for (int t = 0; t < TT; ++t) {
#if defined(HAVE_TDM)
        const int cur = t & 1;
        if (wv == 0 && t + 1 < TT) tdm_issue(t + 1, cur ^ 1);  // overlap with compute
#elif defined(HAVE_ASYNC)
        const int cur = t & 1;
        if (t + 1 < TT) async_issue(t + 1, cur ^ 1);
#else
        const int cur = 0;
        for (int i16 = tid; i16 < 648; i16 += 128) {
            int row = i16 / 72, c16 = i16 % 72;
            int dt = row / 3, dy = row % 3;
            const uint4* src = (const uint4*)(xT +
                ((((long)b * TP + (t + dt)) * HP + (y + dy)) * WP + x0) * CIN) + c16;
            *((uint4*)&sIn[0][row * 576] + c16) = *src;
        }
        __syncthreads();
#endif
        const __bf16* base = &sIn[cur][0];

        // ---- implicit GEMM over the 27 taps, B-frags software-pipelined ----
#define BPTR(qq) (&base[((((qq) / 9) * 3 + ((qq) % 9) / 3) * 18 + (n + (qq) % 3)) * CIN + kB16])
        v8f acc = {0.f, 0.f, 0.f, 0.f, 0.f, 0.f, 0.f, 0.f};
        BF16Frag fbuf[2];
        fbuf[0].q[0] = *(const uint4*)BPTR(0);
        fbuf[0].q[1] = *(const uint4*)(BPTR(0) + 16);
        #pragma unroll
        for (int q = 0; q < 27; ++q) {
            if (q < 26) {
                const __bf16* np = BPTR(q + 1);
                fbuf[(q + 1) & 1].q[0] = *(const uint4*)np;
                fbuf[(q + 1) & 1].q[1] = *(const uint4*)(np + 16);
            }
            BF16Frag a;
            int fidx = ((q * 4 + wv) * 32 + lane) * 2;
            a.q[0] = wq[fidx];
            a.q[1] = wq[fidx + 1];
            acc = __builtin_amdgcn_wmma_f32_16x16x32_bf16(
                false, a.v, false, fbuf[q & 1].v, (short)0, acc, false, false);
        }
#undef BPTR
        #pragma unroll
        for (int r = 0; r < 8; ++r)
            sGates[wv * 16 + r + 8 * (lane >> 4)][n] = acc[r];
        __syncthreads();

        // ---- gating + recurrence: h = f*h + (1-f)*z ----
        #pragma unroll
        for (int k = 0; k < 4; ++k) {
            int i = tid + k * 128;
            int c = i >> 4, xl = i & 15;
            float z  = tanhf(sGates[c][xl] + zb[k]);
            float fg = 1.f / (1.f + expf(-(sGates[c + HID][xl] + fb[k])));
            float h  = fg * sH[c][xl] + (1.f - fg) * z;
            sH[c][xl] = h;
            hT[(((long)(b * HID + c) * TT + t) * HH + y) * WW + x0 + xl] = h;
        }
        __syncthreads();

#if defined(HAVE_TDM)
        if (wv == 0 && t + 1 < TT) WAIT_TENSOR();
        __syncthreads();
#elif defined(HAVE_ASYNC)
        if (t + 1 < TT) WAIT_ASYNC();
        __syncthreads();
#endif
    }
}

// ---------------------------------------------------------------------------
// Kernel 2: per-pixel channel attention, table-order WMMA fragments (k = 8h+s, 16+8h+s-8).
//   GEMM1: scores = qh qh^T from sQ[pix][c][t]
//   softmax over c via in-register + shfl_xor(16) reductions
//   GEMM2: ah^T = qh^T attn^T  from sQt[pix][t][c] and sA[c][d]
// Block = 8 pixels (x), 4 waves x 2 pixels.
// ---------------------------------------------------------------------------
__global__ __launch_bounds__(128) void attn_kernel(
    const float* __restrict__ hT, const float* __restrict__ gptr,
    float* __restrict__ out) {

    __shared__ __align__(16) __bf16 sQ [8][32][32];   // [pix][c][t]
    __shared__ __align__(16) __bf16 sQt[8][32][32];   // [pix][t][c]
    __shared__ __align__(16) __bf16 sA [4][32][32];   // per-wave attn [c][d]

    const int tid  = threadIdx.x;
    const int lane = tid & 31;
    const int w    = tid >> 5;
    const int hi   = lane >> 4;
    const int m    = lane & 15;
    const int kB8  = hi * 8;
    const int kB16 = hi * 16;
    const int x0   = blockIdx.x * 8;
    const int y    = blockIdx.y;
    const int b    = blockIdx.z;
    const float g  = gptr[0];

    for (int idx = tid; idx < 8 * 32 * 32; idx += 128) {
        int pix = idx & 7;
        int t   = (idx >> 3) & 31;
        int c   = idx >> 8;
        float v = 0.f;
        if (t < TT)
            v = hT[(((long)(b * HID + c) * TT + t) * HH + y) * WW + x0 + pix];
        __bf16 bv = (__bf16)v;
        sQ[pix][c][t]  = bv;
        sQt[pix][t][c] = bv;
    }
    __syncthreads();

    for (int pp = 0; pp < 2; ++pp) {
        const int pix = w * 2 + pp;

        // ---- GEMM1: scores (M=c, N=d, K=t padded to 32) ----
        BF16Frag a0, a1, b0, b1;
        a0.q[0] = *(const uint4*)&sQ[pix][m][kB8];
        a0.q[1] = *(const uint4*)&sQ[pix][m][16 + kB8];
        a1.q[0] = *(const uint4*)&sQ[pix][16 + m][kB8];
        a1.q[1] = *(const uint4*)&sQ[pix][16 + m][16 + kB8];
        b0.q[0] = *(const uint4*)&sQ[pix][m][kB8];
        b0.q[1] = *(const uint4*)&sQ[pix][m][16 + kB8];
        b1.q[0] = *(const uint4*)&sQ[pix][16 + m][kB8];
        b1.q[1] = *(const uint4*)&sQ[pix][16 + m][16 + kB8];

        v8f z8 = {0.f, 0.f, 0.f, 0.f, 0.f, 0.f, 0.f, 0.f};
        v8f s00 = __builtin_amdgcn_wmma_f32_16x16x32_bf16(false, a0.v, false, b0.v, (short)0, z8, false, false);
        v8f s01 = __builtin_amdgcn_wmma_f32_16x16x32_bf16(false, a0.v, false, b1.v, (short)0, z8, false, false);
        v8f s10 = __builtin_amdgcn_wmma_f32_16x16x32_bf16(false, a1.v, false, b0.v, (short)0, z8, false, false);
        v8f s11 = __builtin_amdgcn_wmma_f32_16x16x32_bf16(false, a1.v, false, b1.v, (short)0, z8, false, false);

        // ---- softmax over c per column d, then * 1/sqrt(32) ----
        float mx0 = -3.4e38f, mx1 = -3.4e38f;
        #pragma unroll
        for (int r = 0; r < 8; ++r) {
            mx0 = fmaxf(mx0, fmaxf(s00[r], s10[r]));
            mx1 = fmaxf(mx1, fmaxf(s01[r], s11[r]));
        }
        mx0 = fmaxf(mx0, __shfl_xor(mx0, 16, 32));
        mx1 = fmaxf(mx1, __shfl_xor(mx1, 16, 32));
        float sm0 = 0.f, sm1 = 0.f;
        #pragma unroll
        for (int r = 0; r < 8; ++r) {
            s00[r] = expf(s00[r] - mx0); s10[r] = expf(s10[r] - mx0);
            s01[r] = expf(s01[r] - mx1); s11[r] = expf(s11[r] - mx1);
            sm0 += s00[r] + s10[r];
            sm1 += s01[r] + s11[r];
        }
        sm0 += __shfl_xor(sm0, 16, 32);
        sm1 += __shfl_xor(sm1, 16, 32);
        const float sc0 = 0.17677669529663687f / sm0;
        const float sc1 = 0.17677669529663687f / sm1;
        #pragma unroll
        for (int r = 0; r < 8; ++r) {
            sA[w][r + 8 * hi][m]           = (__bf16)(s00[r] * sc0);
            sA[w][16 + r + 8 * hi][m]      = (__bf16)(s10[r] * sc0);
            sA[w][r + 8 * hi][16 + m]      = (__bf16)(s01[r] * sc1);
            sA[w][16 + r + 8 * hi][16 + m] = (__bf16)(s11[r] * sc1);
        }
        __syncthreads();

        // ---- GEMM2: ah^T = qh^T * attn^T  (M=t, N=c, K=d) ----
        BF16Frag e0, e1, f0, f1;
        e0.q[0] = *(const uint4*)&sQt[pix][m][kB8];
        e0.q[1] = *(const uint4*)&sQt[pix][m][16 + kB8];
        e1.q[0] = *(const uint4*)&sQt[pix][16 + m][kB8];
        e1.q[1] = *(const uint4*)&sQt[pix][16 + m][16 + kB8];
        f0.q[0] = *(const uint4*)&sA[w][m][kB8];
        f0.q[1] = *(const uint4*)&sA[w][m][16 + kB8];
        f1.q[0] = *(const uint4*)&sA[w][16 + m][kB8];
        f1.q[1] = *(const uint4*)&sA[w][16 + m][16 + kB8];

        v8f h00 = __builtin_amdgcn_wmma_f32_16x16x32_bf16(false, e0.v, false, f0.v, (short)0, z8, false, false);
        v8f h01 = __builtin_amdgcn_wmma_f32_16x16x32_bf16(false, e0.v, false, f1.v, (short)0, z8, false, false);
        v8f h10 = __builtin_amdgcn_wmma_f32_16x16x32_bf16(false, e1.v, false, f0.v, (short)0, z8, false, false);
        v8f h11 = __builtin_amdgcn_wmma_f32_16x16x32_bf16(false, e1.v, false, f1.v, (short)0, z8, false, false);

        // ---- out = gamma*ah + h_time  (lane holds column c, rows = t) ----
        #pragma unroll
        for (int r = 0; r < 8; ++r) {
            int tA = r + 8 * hi;
            int tB = 16 + tA;
            long iA0 = (((long)(b * HID + m) * TT + tA) * HH + y) * WW + x0 + pix;
            out[iA0] = g * h00[r] + hT[iA0];
            long iA1 = (((long)(b * HID + 16 + m) * TT + tA) * HH + y) * WW + x0 + pix;
            out[iA1] = g * h01[r] + hT[iA1];
            if (tB < TT) {
                long iB0 = (((long)(b * HID + m) * TT + tB) * HH + y) * WW + x0 + pix;
                out[iB0] = g * h10[r] + hT[iB0];
                long iB1 = (((long)(b * HID + 16 + m) * TT + tB) * HH + y) * WW + x0 + pix;
                out[iB1] = g * h11[r] + hT[iB1];
            }
        }
        __syncthreads();
    }
}

// ---------------------------------------------------------------------------
extern "C" void kernel_launch(void* const* d_in, const int* in_sizes, int n_in,
                              void* d_out, int out_size, void* d_ws, size_t ws_size,
                              hipStream_t stream) {
    const float* inp = (const float*)d_in[0];   // [4][32][31][96][96]
    const float* cw  = (const float*)d_in[1];   // [64][32][3][3][3]
    const float* cb  = (const float*)d_in[2];   // [64]
    const float* gm  = (const float*)d_in[3];   // [1]
    float* out = (float*)d_out;

    // workspace: [0,110592) wf bf16 | [128K, +81.2MB) padded bf16 input |
    //            [96MB, +146.3MB) f32 h_time
    __bf16* wf = reinterpret_cast<__bf16*>(d_ws);
    __bf16* xT = reinterpret_cast<__bf16*>((char*)d_ws + 131072);
    float*  hT = reinterpret_cast<float*>((char*)d_ws + 100663296);

    pack_weights<<<(27 * 4 * 32 * 16 + 255) / 256, 256, 0, stream>>>(cw, wf);
    {
        long total = (long)NB * TP * HP * WP * CIN;
        pad_transpose<<<(unsigned)((total + 255) / 256), 256, 0, stream>>>(inp, xT);
    }
    conv_gate_scan<<<dim3(WW / 16, HH, NB), 128, 0, stream>>>(xT, wf, cb, hT);
    attn_kernel<<<dim3(WW / 8, HH, NB), 128, 0, stream>>>(hT, gm, out);
}
